// Extractor_72765335929101
// MI455X (gfx1250) — matrix operand order repaired and checked by reading, not compiled, not run on hardware
//
#include <hip/hip_runtime.h>
#include <hip/hip_bf16.h>

// ---------------- constants (baked from the reference) ----------------
#define BB 4
#define NQ 4096
#define DIM 768
#define HEADS 6
#define DHEAD 128
#define LEVELS 3
#define POINTS 4
#define HID 192
#define NV 21504          // 128*128 + 64*64 + 32*32
#define H0 64
#define W0 64
#define NAW 256           // 144 (off) + 72 (aw) zero-padded to multiple of 64

typedef __attribute__((ext_vector_type(16))) __bf16 v16bf;
typedef __attribute__((ext_vector_type(8)))  float  v8f;

__device__ __constant__ int c_HL[3] = {128, 64, 32};
__device__ __constant__ int c_WL[3] = {128, 64, 32};
__device__ __constant__ int c_ST[3] = {0, 16384, 20480};

// ---------------- LayerNorm over DIM=768, one block per row ----------------
__global__ void ln768_kernel(const float* __restrict__ in,
                             const float* __restrict__ g,
                             const float* __restrict__ b,
                             float* __restrict__ out32,
                             __hip_bfloat16* __restrict__ out16) {
  __shared__ float s1[256];
  __shared__ float s2[256];
  const int tid = threadIdx.x;
  const size_t row = blockIdx.x;
  const float* x = in + row * DIM;
  float l0 = x[tid], l1 = x[tid + 256], l2 = x[tid + 512];
  s1[tid] = l0 + l1 + l2;
  s2[tid] = l0 * l0 + l1 * l1 + l2 * l2;
  __syncthreads();
  for (int off = 128; off > 0; off >>= 1) {
    if (tid < off) { s1[tid] += s1[tid + off]; s2[tid] += s2[tid + off]; }
    __syncthreads();
  }
  const float mean = s1[0] * (1.0f / DIM);
  const float var  = s2[0] * (1.0f / DIM) - mean * mean;
  const float inv  = rsqrtf(var + 1e-6f);
  #pragma unroll
  for (int i = 0; i < 3; ++i) {
    const int c = tid + i * 256;
    const float v = ((i == 0 ? l0 : (i == 1 ? l1 : l2)) - mean) * inv * g[c] + b[c];
    if (out32) out32[row * DIM + c] = v;
    if (out16) out16[row * DIM + c] = __float2bfloat16(v);
  }
}

// ---------------- weight convert + transpose: src[K][N] f32 -> dst[N][K] bf16 ----------------
__global__ void transpose_bf16_kernel(const float* __restrict__ src,
                                      __hip_bfloat16* __restrict__ dst,
                                      int K, int N) {
  const size_t total = (size_t)K * N;
  for (size_t i = (size_t)blockIdx.x * blockDim.x + threadIdx.x; i < total;
       i += (size_t)gridDim.x * blockDim.x) {
    const int n = (int)(i % N);
    const int k = (int)(i / N);
    dst[(size_t)n * K + k] = __float2bfloat16(src[i]);
  }
}

// ---------------- pack off_w|aw_w -> [NAW][768] bf16 (zero padded) + bias ----------------
__global__ void pack_offaw_kernel(const float* __restrict__ off_w,   // [768][144]
                                  const float* __restrict__ aw_w,    // [768][72]
                                  const float* __restrict__ off_b,   // [144]
                                  const float* __restrict__ aw_b,    // [72]
                                  __hip_bfloat16* __restrict__ wT,   // [NAW][768]
                                  float* __restrict__ bias) {        // [NAW]
  const size_t total = (size_t)NAW * DIM;
  const size_t gid0 = (size_t)blockIdx.x * blockDim.x + threadIdx.x;
  for (size_t i = gid0; i < total; i += (size_t)gridDim.x * blockDim.x) {
    const int n = (int)(i / DIM);
    const int k = (int)(i % DIM);
    float v = 0.0f;
    if (n < 144)      v = off_w[(size_t)k * 144 + n];
    else if (n < 216) v = aw_w[(size_t)k * 72 + (n - 144)];
    wT[(size_t)n * DIM + k] = __float2bfloat16(v);
  }
  if (gid0 < NAW) {
    const int n = (int)gid0;
    bias[n] = (n < 144) ? off_b[n] : ((n < 216) ? aw_b[n - 144] : 0.0f);
  }
}

// ---------------- WMMA bf16 GEMM: out = A[M][K] * Bt[N][K]^T + bias (+res1 +res2) ----------------
// 256 threads = 8 waves tiled 4(M) x 2(N); each wave computes 32x32 (4 wmma/K-step,
// double-buffered fragments). Block tile 128x64. M%128==0, N%64==0, K%32==0.
union Frag { uint32_t u[8]; v16bf v; };

#define WMMA_BF16(a, b, c) __builtin_amdgcn_wmma_f32_16x16x32_bf16( \
    false, (a).v, false, (b).v, (short)0, (c), false, false)

__global__ void gemm_wmma_bf16_kernel(const __hip_bfloat16* __restrict__ A,
                                      const __hip_bfloat16* __restrict__ Bt,
                                      const float* __restrict__ bias,
                                      const float* __restrict__ res1,
                                      const float* __restrict__ res2,
                                      float* __restrict__ out32,
                                      __hip_bfloat16* __restrict__ out16,
                                      int M, int N, int K) {
  const int lane = threadIdx.x & 31;
  const int wave = threadIdx.x >> 5;
  const int wm = wave & 3;          // 4 waves along M
  const int wn = wave >> 2;         // 2 waves along N
  const int r = lane & 15;
  const int half = lane >> 4;

  const int tm = blockIdx.x * 128 + wm * 32;  // wave covers rows [tm, tm+32)
  const int tn = blockIdx.y * 64 + wn * 32;   // wave covers cols [tn, tn+32)

  const size_t arow0 = (size_t)(tm + r) * K;
  const size_t arow1 = (size_t)(tm + 16 + r) * K;
  const size_t bcol0 = (size_t)(tn + r) * K;
  const size_t bcol1 = (size_t)(tn + 16 + r) * K;

  // 16-bit A 16x32 VGPR layout: regs 0..3 hold K = half*8 + {0..7},
  // regs 4..7 hold K = 16 + half*8 + {0..7}  (pairs of bf16 per reg)
  const int ka_lo = half * 8;
  // B 32x16 layout: same K order as A (regs 0..3 K = half*8 + {0..7}, regs 4..7 K = 16 + half*8 + {0..7})
  const int kb_lo = half * 8;

  auto loadA = [&](size_t base, int k0, Frag& f) {
    #pragma unroll
    for (int vv = 0; vv < 8; ++vv) {
      const int ka = (vv < 4) ? (ka_lo + vv * 2) : (16 + ka_lo + (vv - 4) * 2);
      f.u[vv] = *reinterpret_cast<const uint32_t*>(A + base + k0 + ka);
    }
  };
  auto loadB = [&](size_t base, int k0, Frag& f) {
    #pragma unroll
    for (int vv = 0; vv < 8; ++vv) {
      f.u[vv] = *reinterpret_cast<const uint32_t*>(Bt + base + k0 + ((vv < 4) ? (kb_lo + vv * 2) : (16 + kb_lo + (vv - 4) * 2)));
    }
  };

  v8f acc00 = {}, acc01 = {}, acc10 = {}, acc11 = {};
  Frag a0c, a1c, b0c, b1c, a0n, a1n, b0n, b1n;

  loadA(arow0, 0, a0c); loadA(arow1, 0, a1c);
  loadB(bcol0, 0, b0c); loadB(bcol1, 0, b1c);

  for (int k0 = 32; k0 < K; k0 += 32) {
    if (k0 + 32 < K) {
      __builtin_prefetch(A + arow0 + k0 + 32, 0, 0);
      __builtin_prefetch(A + arow1 + k0 + 32, 0, 0);
    }
    // issue next-step loads before this step's wmmas (double buffer)
    loadA(arow0, k0, a0n); loadA(arow1, k0, a1n);
    loadB(bcol0, k0, b0n); loadB(bcol1, k0, b1n);
    acc00 = WMMA_BF16(a0c, b0c, acc00);
    acc01 = WMMA_BF16(a0c, b1c, acc01);
    acc10 = WMMA_BF16(a1c, b0c, acc10);
    acc11 = WMMA_BF16(a1c, b1c, acc11);
    a0c = a0n; a1c = a1n; b0c = b0n; b1c = b1n;
  }
  acc00 = WMMA_BF16(a0c, b0c, acc00);
  acc01 = WMMA_BF16(a0c, b1c, acc01);
  acc10 = WMMA_BF16(a1c, b0c, acc10);
  acc11 = WMMA_BF16(a1c, b1c, acc11);

  const float bv0 = bias ? bias[tn + r] : 0.0f;
  const float bv1 = bias ? bias[tn + 16 + r] : 0.0f;
  #pragma unroll
  for (int mi = 0; mi < 2; ++mi) {
    #pragma unroll
    for (int ni = 0; ni < 2; ++ni) {
      const v8f acc = (mi == 0) ? (ni == 0 ? acc00 : acc01) : (ni == 0 ? acc10 : acc11);
      const int col = tn + ni * 16 + r;
      const float bv = ni == 0 ? bv0 : bv1;
      #pragma unroll
      for (int j = 0; j < 8; ++j) {
        const int row = tm + mi * 16 + half * 8 + j;  // C/D: reg j -> M=j (+8 high half)
        const size_t idx = (size_t)row * N + col;
        float v = acc[j] + bv;
        if (res1) v += res1[idx];
        if (res2) v += res2[idx];
        if (out32) out32[idx] = v;
        if (out16) out16[idx] = __float2bfloat16(v);
      }
    }
  }
}

// ---------------- softmax(aw) + bilinear sampling + weighted sum ----------------
// one block per (b, nq, head); 128 threads = channels of that head
__global__ void sample_kernel(const float* __restrict__ offaw,   // [B*NQ][NAW]
                              const float* __restrict__ ref,     // [B][NQ][3][2]
                              const __hip_bfloat16* __restrict__ vbf, // [B][NV][768]
                              __hip_bfloat16* __restrict__ samp) {    // [B][NQ][768]
  const int head = blockIdx.x % HEADS;
  const int bq = blockIdx.x / HEADS;          // b*NQ + nq
  const int b = bq / NQ;
  const int tid = threadIdx.x;                // 0..127 channel within head

  const float* row = offaw + (size_t)bq * NAW;

  // softmax over the 12 attention logits of this head
  float e[LEVELS * POINTS];
  float mx = -1e30f;
  #pragma unroll
  for (int i = 0; i < LEVELS * POINTS; ++i) {
    e[i] = row[144 + head * (LEVELS * POINTS) + i];
    mx = fmaxf(mx, e[i]);
  }
  float denom = 0.0f;
  #pragma unroll
  for (int i = 0; i < LEVELS * POINTS; ++i) { e[i] = __expf(e[i] - mx); denom += e[i]; }
  const float rdenom = 1.0f / denom;

  float acc = 0.0f;
  #pragma unroll
  for (int l = 0; l < LEVELS; ++l) {
    const int Hl = c_HL[l], Wl = c_WL[l];
    const int start = c_ST[l];
    const float rx = ref[((size_t)bq * LEVELS + l) * 2 + 0];
    const float ry = ref[((size_t)bq * LEVELS + l) * 2 + 1];
    #pragma unroll
    for (int p = 0; p < POINTS; ++p) {
      const int oc = ((head * LEVELS + l) * POINTS + p) * 2;
      const float lx = rx + row[oc + 0] / (float)Wl;
      const float ly = ry + row[oc + 1] / (float)Hl;
      // grid = 2*loc-1; x = (g+1)*W/2-0.5 == loc*W - 0.5
      const float xx = lx * (float)Wl - 0.5f;
      const float yy = ly * (float)Hl - 0.5f;
      const float x0f = floorf(xx), y0f = floorf(yy);
      const int x0 = (int)x0f, y0 = (int)y0f;
      const float wx = xx - x0f, wy = yy - y0f;
      const float aw = e[l * POINTS + p] * rdenom;
      #pragma unroll
      for (int cy = 0; cy < 2; ++cy) {
        #pragma unroll
        for (int cx = 0; cx < 2; ++cx) {
          const int xi = x0 + cx, yi = y0 + cy;
          if (xi >= 0 && xi < Wl && yi >= 0 && yi < Hl) {
            const float w = aw * (cx ? wx : (1.0f - wx)) * (cy ? wy : (1.0f - wy));
            const size_t vidx =
                ((size_t)b * NV + start + (size_t)yi * Wl + xi) * DIM + head * DHEAD + tid;
            acc += w * __bfloat162float(vbf[vidx]);
          }
        }
      }
    }
  }
  samp[(size_t)bq * DIM + head * DHEAD + tid] = __float2bfloat16(acc);
}

// ---------------- depthwise 3x3 conv (SAME, zero pad) + exact GELU ----------------
// one block per (b, h, w); HID=192 threads = channels
__global__ void dwconv_gelu_kernel(const float* __restrict__ y1,   // [B][H][W][HID]
                                   const float* __restrict__ dw_w, // [3][3][1][HID]
                                   const float* __restrict__ dw_b, // [HID]
                                   __hip_bfloat16* __restrict__ y2) {
  const int c = threadIdx.x;
  const int w = blockIdx.x % W0;
  const int h = (blockIdx.x / W0) % H0;
  const int b = blockIdx.x / (H0 * W0);
  float acc = dw_b[c];
  #pragma unroll
  for (int ky = 0; ky < 3; ++ky) {
    const int hh = h + ky - 1;
    if (hh < 0 || hh >= H0) continue;
    #pragma unroll
    for (int kx = 0; kx < 3; ++kx) {
      const int ww = w + kx - 1;
      if (ww < 0 || ww >= W0) continue;
      acc += y1[(((size_t)b * H0 + hh) * W0 + ww) * HID + c] * dw_w[(ky * 3 + kx) * HID + c];
    }
  }
  const float g = 0.5f * acc * (1.0f + erff(acc * 0.70710678118654752f));
  y2[(size_t)blockIdx.x * HID + c] = __float2bfloat16(g);
}

// ---------------- launch ----------------
extern "C" void kernel_launch(void* const* d_in, const int* in_sizes, int n_in,
                              void* d_out, int out_size, void* d_ws, size_t ws_size,
                              hipStream_t stream) {
  const float* query = (const float*)d_in[0];
  const float* refpt = (const float*)d_in[1];
  const float* feat  = (const float*)d_in[2];
  const float* qn_g  = (const float*)d_in[7];
  const float* qn_b  = (const float*)d_in[8];
  const float* fn_g  = (const float*)d_in[9];
  const float* fn_b  = (const float*)d_in[10];
  const float* off_w = (const float*)d_in[11];
  const float* off_b = (const float*)d_in[12];
  const float* aw_w  = (const float*)d_in[13];
  const float* aw_b  = (const float*)d_in[14];
  const float* vp_w  = (const float*)d_in[15];
  const float* vp_b  = (const float*)d_in[16];
  const float* op_w  = (const float*)d_in[17];
  const float* op_b  = (const float*)d_in[18];
  const float* ffn_g = (const float*)d_in[19];
  const float* ffn_b = (const float*)d_in[20];
  const float* fc1_w = (const float*)d_in[21];
  const float* fc1_b = (const float*)d_in[22];
  const float* dw_w  = (const float*)d_in[23];
  const float* dw_b  = (const float*)d_in[24];
  const float* fc2_w = (const float*)d_in[25];
  const float* fc2_b = (const float*)d_in[26];
  float* outp = (float*)d_out;

  const size_t MQ = (size_t)BB * NQ;   // 16384 query rows
  const size_t MV = (size_t)BB * NV;   // 86016 value rows

  // ---- workspace carve-out ----
  size_t off = 0;
  auto alloc = [&](size_t bytes) -> char* {
    char* p = (char*)d_ws + off;
    off = (off + bytes + 255) & ~(size_t)255;
    return p;
  };
  float*          q32     = (float*)alloc(MQ * DIM * sizeof(float));
  __hip_bfloat16* qbf     = (__hip_bfloat16*)alloc(MQ * DIM * 2);
  __hip_bfloat16* fnbf    = (__hip_bfloat16*)alloc(MV * DIM * 2);
  __hip_bfloat16* vbf     = (__hip_bfloat16*)alloc(MV * DIM * 2);
  float*          offaw   = (float*)alloc(MQ * NAW * sizeof(float));
  __hip_bfloat16* sampbf  = (__hip_bfloat16*)alloc(MQ * DIM * 2);
  float*          x32     = (float*)alloc(MQ * DIM * sizeof(float));
  __hip_bfloat16* lnxbf   = (__hip_bfloat16*)alloc(MQ * DIM * 2);
  float*          y1      = (float*)alloc(MQ * HID * sizeof(float));
  __hip_bfloat16* y2bf    = (__hip_bfloat16*)alloc(MQ * HID * 2);
  __hip_bfloat16* vpT     = (__hip_bfloat16*)alloc((size_t)DIM * DIM * 2);
  __hip_bfloat16* opT     = (__hip_bfloat16*)alloc((size_t)DIM * DIM * 2);
  __hip_bfloat16* offawT  = (__hip_bfloat16*)alloc((size_t)NAW * DIM * 2);
  __hip_bfloat16* fc1T    = (__hip_bfloat16*)alloc((size_t)HID * DIM * 2);
  __hip_bfloat16* fc2T    = (__hip_bfloat16*)alloc((size_t)DIM * HID * 2);
  float*          oab     = (float*)alloc(NAW * sizeof(float));
  (void)ws_size;

  // ---- weight prep ----
  transpose_bf16_kernel<<<2304, 256, 0, stream>>>(vp_w, vpT, DIM, DIM);
  transpose_bf16_kernel<<<2304, 256, 0, stream>>>(op_w, opT, DIM, DIM);
  transpose_bf16_kernel<<<576, 256, 0, stream>>>(fc1_w, fc1T, DIM, HID);
  transpose_bf16_kernel<<<576, 256, 0, stream>>>(fc2_w, fc2T, HID, DIM);
  pack_offaw_kernel<<<768, 256, 0, stream>>>(off_w, aw_w, off_b, aw_b, offawT, oab);

  // ---- layernorms of inputs ----
  ln768_kernel<<<(int)MQ, 256, 0, stream>>>(query, qn_g, qn_b, q32, qbf);
  ln768_kernel<<<(int)MV, 256, 0, stream>>>(feat, fn_g, fn_b, nullptr, fnbf);

  // ---- v = LN(feat) @ vp_w + vp_b  (-> bf16) ----
  gemm_wmma_bf16_kernel<<<dim3((int)(MV / 128), DIM / 64), 256, 0, stream>>>(
      fnbf, vpT, vp_b, nullptr, nullptr, nullptr, vbf, (int)MV, DIM, DIM);

  // ---- off/aw logits = q @ [off_w|aw_w] + bias ----
  gemm_wmma_bf16_kernel<<<dim3((int)(MQ / 128), NAW / 64), 256, 0, stream>>>(
      qbf, offawT, oab, nullptr, nullptr, offaw, nullptr, (int)MQ, NAW, DIM);

  // ---- softmax + bilinear sampling ----
  sample_kernel<<<(int)(MQ * HEADS), DHEAD, 0, stream>>>(offaw, refpt, vbf, sampbf);

  // ---- x = samp @ op_w + op_b + q + query ----
  gemm_wmma_bf16_kernel<<<dim3((int)(MQ / 128), DIM / 64), 256, 0, stream>>>(
      sampbf, opT, op_b, q32, query, x32, nullptr, (int)MQ, DIM, DIM);

  // ---- LN(x) -> bf16 ----
  ln768_kernel<<<(int)MQ, 256, 0, stream>>>(x32, ffn_g, ffn_b, nullptr, lnxbf);

  // ---- y1 = LN(x) @ fc1_w + fc1_b ----
  gemm_wmma_bf16_kernel<<<dim3((int)(MQ / 128), HID / 64), 256, 0, stream>>>(
      lnxbf, fc1T, fc1_b, nullptr, nullptr, y1, nullptr, (int)MQ, HID, DIM);

  // ---- depthwise 3x3 + GELU ----
  dwconv_gelu_kernel<<<(int)MQ, HID, 0, stream>>>(y1, dw_w, dw_b, y2bf);

  // ---- out = x + (gelu @ fc2_w + fc2_b) ----
  gemm_wmma_bf16_kernel<<<dim3((int)(MQ / 128), DIM / 64), 256, 0, stream>>>(
      y2bf, fc2T, fc2_b, x32, nullptr, outp, nullptr, (int)MQ, DIM, HID);
  (void)in_sizes; (void)n_in; (void)out_size;
}
